// DRKXXJ_27178553049145
// MI455X (gfx1250) — compile-verified, matrix operand order checked by reading, not run on hardware
//
#include <hip/hip_runtime.h>
#include <hip/hip_bf16.h>
#include <math.h>

// ---------------------------------------------------------------------------
// MI455X (gfx1250) implementation. Wave32; WMMA f32<-f16 16x16x32 for all
// GEMMs (dense convs via implicit im2col, Mamba projections, heads).
// Weights pre-packed to the WMMA fragment layout (f16, 32x16 tiles); B tiles
// staged with global_load_async_to_lds_b128 (ASYNCcnt); fragments loaded as
// contiguous 32B per lane (2x ds_load_b128).
// ---------------------------------------------------------------------------

typedef __attribute__((ext_vector_type(16))) _Float16 v16h;
typedef __attribute__((ext_vector_type(8)))  float    v8f;

enum { ACT_NONE = 0, ACT_GELU = 1, ACT_SILU = 2, ACT_SOFTPLUS = 3 };

__device__ __forceinline__ float gelu_f(float x) {
  float x3 = x * x * x;
  return 0.5f * x * (1.f + tanhf(0.7978845608028654f * (x + 0.044715f * x3)));
}
__device__ __forceinline__ float silu_f(float x) { return x / (1.f + expf(-x)); }
__device__ __forceinline__ float softplus_f(float x) {
  return (x > 20.f) ? x : log1pf(expf(x));
}

// ---------------------------------------------------------------------------
// Weight pre-pack: f32 row-major (K x N) -> f16 WMMA fragment tiles
//   out[kt][nt][lane][slot], kt = k/32, nt = n/16,
//   lane = (n&15) | (((k>>3)&1)<<4),  slot = (k&7) | ((k&16)>>1)
// Zero-padded to full tiles, so GEMM staging needs no bounds checks.
// ---------------------------------------------------------------------------
__global__ void k_pack_b(const float* __restrict__ in, _Float16* __restrict__ out,
                         int K, int N, int KT, int NT)
{
  long i = (long)blockIdx.x * blockDim.x + threadIdx.x;
  long total = (long)KT * NT * 512;
  if (i >= total) return;
  int within = (int)(i & 511);
  long tile  = i >> 9;
  int nt = (int)(tile % NT);
  int kt = (int)(tile / NT);
  int lane = within >> 4;
  int s    = within & 15;
  int n = nt * 16 + (lane & 15);
  int k = kt * 32 + ((s & 7) | ((s & 8) << 1) | ((lane >> 4) << 3));
  float v = (k < K && n < N) ? in[(long)k * N + n] : 0.f;
  out[i] = (_Float16)v;
}

// ---------------------------------------------------------------------------
// WMMA GEMM:  C[M,N] = act(A[M,K] * B16[K,N] + bias) (+ res)
//   - A: f32, plain row-major (cin==0, leading dim lda) or implicit im2col of
//     a length-M, cin-channel sequence (kernel ks, left pad `pad`, K=ks*cin).
//   - Bp: f16 pre-packed fragment tiles (see k_pack_b), NT tiles per K-row.
// Block = 256 threads = 8 waves. Block tile 128(M) x 64(N), K step 32.
// Each wave owns a 16x64 strip: 1 A fragment reused over 4 B fragments ->
// 4 x v_wmma_f32_16x16x32_f16 per K step. All fragment LDS traffic is
// 2 x ds_load_b128 per fragment. B staged via GLOBAL_LOAD_ASYNC_TO_LDS_B128.
// ---------------------------------------------------------------------------
#define BM 128
#define BN 64
#define TKK 32

__global__ __launch_bounds__(256) void k_gemm(
    const float* __restrict__ A, const _Float16* __restrict__ Bp,
    const float* __restrict__ bias, const float* __restrict__ res,
    float* __restrict__ C,
    int M, int N, int K, int lda, int ldc,
    int act, int cin, int ks, int pad)
{
  __shared__ __align__(32) _Float16 Ap[8][512];   // 8 subtiles x (32 lanes x 16)
  __shared__ __align__(32) _Float16 Bq[4][512];   // 4 subtiles x (32 lanes x 16)

  const int tid  = threadIdx.x;
  const int m0   = blockIdx.y * BM;
  const int n0   = blockIdx.x * BN;
  const int wave = tid >> 5;
  const int lane = tid & 31;
  const int h    = lane >> 4;
  const int l15  = lane & 15;

  const int KT  = (K + 31) >> 5;
  const int NT  = (N + 15) >> 4;
  const int nt0 = n0 >> 4;

  v8f acc[4] = {v8f{}, v8f{}, v8f{}, v8f{}};

  for (int kt = 0; kt < KT; ++kt) {
    const int kk = kt << 5;
    // ---- stage A tile (128 rows x 32 k), f32 -> f16, permuted layout ----
    {
      int r     = tid >> 1;                 // 0..127
      int halfk = tid & 1;                  // k group: 0..15 / 16..31
      int m     = m0 + r;
      _Float16* sub = Ap[r >> 4];
#pragma unroll
      for (int g = 0; g < 2; ++g) {
        int kbase = (halfk << 4) + (g << 3);          // 0,8,16,24
        int lane_ = (r & 15) | (((kbase >> 3) & 1) << 4);
        int sbase = (kbase & 16) >> 1;                // 0 or 8
        _Float16* dst = sub + lane_ * 16 + sbase;
#pragma unroll
        for (int i = 0; i < 8; ++i) {
          int k = kk + kbase + i;
          float v = 0.f;
          if (m < M && k < K) {
            if (cin > 0) {                  // im2col: K = ks*cin, SAME conv
              int dk  = k / cin;
              int ci  = k - dk * cin;
              int src = m + dk - pad;
              if (src >= 0 && src < M) v = A[(long)src * cin + ci];
            } else {
              v = A[(long)m * lda + k];
            }
          }
          dst[i] = (_Float16)v;
        }
      }
    }
    // ---- stage B tile: 4 packed 1KB subtiles, async global -> LDS ----
    {
      int j  = tid >> 6;                    // n-subtile 0..3
      int c8 = (tid & 63) << 3;             // 8-f16 chunk within subtile
      _Float16* lq = &Bq[0][0] + (tid << 3);
      if (nt0 + j < NT) {
        unsigned long long gp = (unsigned long long)(
            Bp + (((long)kt * NT + nt0 + j) << 9) + c8);
        unsigned lp = (unsigned)(size_t)lq;
        asm volatile("global_load_async_to_lds_b128 %0, %1, off"
                     :: "v"(lp), "v"(gp) : "memory");
      } else {
#pragma unroll
        for (int i = 0; i < 8; ++i) lq[i] = (_Float16)0.f;
      }
      asm volatile("s_wait_asynccnt 0x0" ::: "memory");
      if (kt + 1 < KT) {                    // gfx1250 global_prefetch next K tile
        __builtin_prefetch(Bp + (((long)(kt + 1) * NT + nt0) << 9) + (tid << 3),
                           0, 3);
      }
    }
    __syncthreads();

    // ---- fragments: one 32B contiguous LDS read per lane (2x b128) ----
    v16h af = *(const v16h*)(Ap[wave] + (lane << 4));
#pragma unroll
    for (int j = 0; j < 4; ++j) {
      v16h bf = *(const v16h*)(Bq[j] + (lane << 4));
      acc[j] = __builtin_amdgcn_wmma_f32_16x16x32_f16(false, af, false, bf,
                                                      (short)0, acc[j],
                                                      false, false);
    }
    __syncthreads();
  }

  // ---- epilogue: bias + activation (+ residual), per C/D VGPR layout ----
  {
    int mb = h << 3;
#pragma unroll
    for (int j = 0; j < 4; ++j) {
      int n = n0 + (j << 4) + l15;
#pragma unroll
      for (int v = 0; v < 8; ++v) {
        int m = m0 + (wave << 4) + mb + v;
        if (m < M && n < N) {
          float val = acc[j][v];
          if (bias) val += bias[n];
          if (act == ACT_GELU)          val = gelu_f(val);
          else if (act == ACT_SILU)     val = silu_f(val);
          else if (act == ACT_SOFTPLUS) val = softplus_f(val);
          if (res) val += res[(long)m * ldc + n];
          C[(long)m * ldc + n] = val;
        }
      }
    }
  }
}

// ---------------------------------------------------------------------------
// LayerNorm over feature dim (optionally fused GELU). One block per row.
// ---------------------------------------------------------------------------
__global__ __launch_bounds__(256) void k_layernorm(
    const float* __restrict__ x, const float* __restrict__ g,
    const float* __restrict__ b, float* __restrict__ out, int Cf, int do_gelu)
{
  __shared__ float red[256];
  long row = blockIdx.x;
  const float* xr = x + row * Cf;
  float s = 0.f;
  for (int c = threadIdx.x; c < Cf; c += 256) s += xr[c];
  red[threadIdx.x] = s; __syncthreads();
  for (int o = 128; o > 0; o >>= 1) {
    if (threadIdx.x < o) red[threadIdx.x] += red[threadIdx.x + o];
    __syncthreads();
  }
  float mean = red[0] / (float)Cf;
  __syncthreads();
  float sq = 0.f;
  for (int c = threadIdx.x; c < Cf; c += 256) { float d = xr[c] - mean; sq += d * d; }
  red[threadIdx.x] = sq; __syncthreads();
  for (int o = 128; o > 0; o >>= 1) {
    if (threadIdx.x < o) red[threadIdx.x] += red[threadIdx.x + o];
    __syncthreads();
  }
  float rstd = rsqrtf(red[0] / (float)Cf + 1e-6f);
  for (int c = threadIdx.x; c < Cf; c += 256) {
    float v = (xr[c] - mean) * rstd * g[c] + b[c];
    out[row * Cf + c] = do_gelu ? gelu_f(v) : v;
  }
}

// maxpool2 along sequence: out[p,c] = max(in[2p,c], in[2p+1,c])
__global__ void k_maxpool2(const float* __restrict__ in, float* __restrict__ out,
                           long Lout, int Cf)
{
  long i = (long)blockIdx.x * blockDim.x + threadIdx.x;
  if (i >= Lout * Cf) return;
  int  c = (int)(i % Cf);
  long p = i / Cf;
  out[i] = fmaxf(in[(2 * p) * Cf + c], in[(2 * p + 1) * Cf + c]);
}

// causal depthwise conv (K=4) + SiLU; rev flips the input sequence
__global__ void k_dwconv_silu(const float* __restrict__ u, const float* __restrict__ w,
                              const float* __restrict__ bias, float* __restrict__ uc,
                              int LTn, int Ef, int rev)
{
  long i = (long)blockIdx.x * blockDim.x + threadIdx.x;
  if (i >= (long)LTn * Ef) return;
  int e = (int)(i % Ef);
  int t = (int)(i / Ef);
  float acc = bias[e];
#pragma unroll
  for (int k = 0; k < 4; ++k) {
    int tt = t - 3 + k;
    if (tt >= 0) {
      int src = rev ? (LTn - 1 - tt) : tt;
      acc += u[(long)src * Ef + e] * w[k * Ef + e];
    }
  }
  uc[i] = silu_f(acc);
}

// selective scan: one thread per channel e, 8 states in registers.
__global__ void k_scan(const float* __restrict__ uc, const float* __restrict__ xdbc,
                       const float* __restrict__ dt, const float* __restrict__ A_log,
                       const float* __restrict__ Dp, const float* __restrict__ z,
                       float* __restrict__ ysum, int LTn, int Ef, int rev, int accum)
{
  int e = blockIdx.x * blockDim.x + threadIdx.x;
  if (e >= Ef) return;
  float An[8];
#pragma unroll
  for (int n = 0; n < 8; ++n) An[n] = -expf(A_log[e * 8 + n]);
  float hs[8] = {0, 0, 0, 0, 0, 0, 0, 0};
  float Dv = Dp[e];
  for (int t = 0; t < LTn; ++t) {
    float dtv = dt[(long)t * Ef + e];
    float ucv = uc[(long)t * Ef + e];
    const float* xr = xdbc + (long)t * 64;   // [48:56]=B, [56:64]=C
    float acc = 0.f;
#pragma unroll
    for (int n = 0; n < 8; ++n) {
      float a = expf(dtv * An[n]);
      hs[n] = a * hs[n] + dtv * ucv * xr[48 + n];
      acc += hs[n] * xr[56 + n];
    }
    float y = acc + ucv * Dv;
    int tn = rev ? (LTn - 1 - t) : t;
    y *= silu_f(z[(long)tn * Ef + e]);
    if (accum) ysum[(long)tn * Ef + e] += y;
    else       ysum[(long)tn * Ef + e]  = y;
  }
}

// gather [xd, xa] pairs into G (J x 2D)
__global__ void k_gather(const float* __restrict__ xc, const int* __restrict__ idx,
                         float* __restrict__ G, int Dfeat)
{
  int j  = blockIdx.x;
  int i0 = idx[2 * j], i1 = idx[2 * j + 1];
  for (int c = threadIdx.x; c < Dfeat; c += blockDim.x) {
    G[(long)j * 2 * Dfeat + c]          = xc[(long)i0 * Dfeat + c];
    G[(long)j * 2 * Dfeat + Dfeat + c]  = xc[(long)i1 * Dfeat + c];
  }
}

// final 64 -> 1 + softplus
__global__ void k_head2(const float* __restrict__ H, const float* __restrict__ w2,
                        const float* __restrict__ b2, float* __restrict__ out,
                        int Jn, int Ff)
{
  int j = blockIdx.x * blockDim.x + threadIdx.x;
  if (j >= Jn) return;
  float a = b2[0];
  for (int f = 0; f < Ff; ++f) a += H[(long)j * Ff + f] * w2[f];
  out[j] = softplus_f(a);
}

// ---------------------------------------------------------------------------
// host-side launcher
// ---------------------------------------------------------------------------
static inline void gemm(hipStream_t s, const float* A, const _Float16* Bp,
                        const float* bias, const float* res, float* C,
                        int M, int N, int K, int lda, int ldc,
                        int act, int cin = 0, int ks = 0, int pad = 0)
{
  dim3 g((N + BN - 1) / BN, (M + BM - 1) / BM);
  k_gemm<<<g, 256, 0, s>>>(A, Bp, bias, res, C, M, N, K, lda, ldc,
                           act, cin, ks, pad);
}

extern "C" void kernel_launch(void* const* d_in, const int* in_sizes, int n_in,
                              void* d_out, int out_size, void* d_ws, size_t ws_size,
                              hipStream_t stream)
{
  (void)in_sizes; (void)n_in; (void)out_size; (void)ws_size;
  const int L = 196608, D = 768, E = 576, LT = 6144, CROP = 2048, LC = 2048;
  const int Jn = 1024, F = 64;

  const float* x_in = (const float*)d_in[0];
  const int*   sidx = (const int*)d_in[1];
  auto P = [&](int i) { return (const float*)d_in[i]; };

  // workspace arena
  char* ws = (char*)d_ws;
  float* BUF_P = (float*)(ws);                              // 256 MB
  float* BUF_Q = (float*)(ws + ((size_t)256 << 20));        // 160 MB
  float* BUF_R = (float*)(ws + ((size_t)416 << 20));        // 160 MB
  char*  S     = (ws + ((size_t)576 << 20));                // 64 MB scratch
  float* uc   = (float*)(S);
  float* xdbc = (float*)(S + ((size_t)16 << 20));
  float* dtb  = (float*)(S + ((size_t)18 << 20));
  float* ysum = (float*)(S + ((size_t)34 << 20));
  float* Gbuf = (float*)(S);                  // reused post-mamba
  float* Hbuf = (float*)(S + ((size_t)8 << 20));

  // ---- pack all GEMM weights into f16 WMMA fragment tiles ----
  _Float16* w16base = (_Float16*)(ws + ((size_t)640 << 20));   // 32 MB region
  size_t wofs = 0;
  auto pack = [&](int i, int K, int N) -> const _Float16* {
    int KT = (K + 31) >> 5, NT = (N + 15) >> 4;
    long total = (long)KT * NT * 512;
    _Float16* dst = w16base + wofs;
    wofs += (size_t)total;
    k_pack_b<<<(unsigned)((total + 255) / 256), 256, 0, stream>>>(
        (const float*)d_in[i], dst, K, N, KT, NT);
    return dst;
  };
  const _Float16* dnaW = pack(2, 36, 320);
  const int feats[5] = {320, 384, 483, 609, 768};
  const _Float16* towerW[4];
  for (int i = 0; i < 4; ++i) towerW[i] = pack(6 + 4 * i, 5 * feats[i], feats[i + 1]);
  const _Float16 *winW[6], *wzW[6], *woutW[6], *xprojW[6][2], *dtprojW[6][2];
  for (int l = 0; l < 6; ++l) {
    int base = 20 + 19 * l;
    winW[l]  = pack(base + 2, D, E);
    wzW[l]   = pack(base + 3, D, E);
    woutW[l] = pack(base + 4, E, D);
    for (int d = 0; d < 2; ++d) {
      int o = base + 5 + 7 * d;
      xprojW[l][d]  = pack(o + 2, E, 64);
      dtprojW[l][d] = pack(o + 3, 48, E);
    }
  }
  const _Float16* finalW = pack(134, D, F);
  const _Float16* xxjW1  = pack(136, 2 * D, 64);

  // ---------------- trunk ----------------
  gemm(stream, x_in, dnaW, P(3), nullptr, BUF_P, L, 320, 36, 0, 320,
       ACT_NONE, /*cin*/4, 9, 4);
  {
    long n = (long)(L / 2) * 320;
    k_maxpool2<<<(unsigned)((n + 255) / 256), 256, 0, stream>>>(BUF_P, BUF_Q, L / 2, 320);
  }
  int len = L / 2;
  for (int i = 0; i < 4; ++i) {
    int cf = feats[i], co = feats[i + 1];
    k_layernorm<<<len, 256, 0, stream>>>(BUF_Q, P(4 + 4 * i), P(5 + 4 * i),
                                         BUF_R, cf, /*gelu*/1);
    gemm(stream, BUF_R, towerW[i], P(7 + 4 * i), nullptr, BUF_P,
         len, co, 5 * cf, 0, co, ACT_NONE, cf, 5, 2);
    long n = (long)(len / 2) * co;
    k_maxpool2<<<(unsigned)((n + 255) / 256), 256, 0, stream>>>(BUF_P, BUF_Q, len / 2, co);
    len /= 2;
  }
  // x lives in BUF_Q : (6144, 768)

  // ---------------- bidirectional Mamba x6 ----------------
  float* xn   = BUF_R;
  float* u    = (float*)((char*)BUF_R + ((size_t)20 << 20));
  float* zb   = (float*)((char*)BUF_R + ((size_t)40 << 20));
  float* xalt = (float*)((char*)BUF_R + ((size_t)60 << 20));
  float* xcur = BUF_Q;

  for (int l = 0; l < 6; ++l) {
    int base = 20 + 19 * l;
    float* xout = (xcur == BUF_Q) ? xalt : BUF_Q;

    k_layernorm<<<LT, 256, 0, stream>>>(xcur, P(base), P(base + 1), xn, D, 0);
    gemm(stream, xn, winW[l], nullptr, nullptr, u,  LT, E, D, D, E, ACT_NONE);
    gemm(stream, xn, wzW[l],  nullptr, nullptr, zb, LT, E, D, D, E, ACT_NONE);

    for (int d = 0; d < 2; ++d) {
      int o = base + 5 + 7 * d;   // conv_w, conv_b, xproj, dtproj, dt_bias, A_log, D
      long n = (long)LT * E;
      k_dwconv_silu<<<(unsigned)((n + 255) / 256), 256, 0, stream>>>(
          u, P(o), P(o + 1), uc, LT, E, d);
      gemm(stream, uc, xprojW[l][d], nullptr, nullptr, xdbc,
           LT, 64, E, E, 64, ACT_NONE);
      gemm(stream, xdbc, dtprojW[l][d], P(o + 4), nullptr, dtb,
           LT, E, 48, 64, E, ACT_SOFTPLUS);
      k_scan<<<(E + 63) / 64, 64, 0, stream>>>(uc, xdbc, dtb, P(o + 5), P(o + 6),
                                               zb, ysum, LT, E, d, d);
    }
    gemm(stream, ysum, woutW[l], nullptr, xcur, xout, LT, D, E, E, D, ACT_NONE);
    xcur = xout;
  }
  // after 6 layers xcur == BUF_Q

  // ---------------- heads ----------------
  const float* xc = xcur + (long)CROP * D;        // cropped (2048, 768)
  float* out_f = (float*)d_out;

  gemm(stream, xc, finalW, P(135), nullptr, out_f, LC, F, D, D, F, ACT_SOFTPLUS);

  k_gather<<<Jn, 256, 0, stream>>>(xc, sidx, Gbuf, D);
  gemm(stream, Gbuf, xxjW1, P(137), nullptr, Hbuf, Jn, 64, 2 * D, 2 * D, 64,
       ACT_GELU);
  k_head2<<<(Jn + 255) / 256, 256, 0, stream>>>(Hbuf, P(138), P(139),
                                                out_f + (long)LC * F, Jn, 64);
}
